// StatePeriDynamicAttention_43980465111631
// MI455X (gfx1250) — matrix operand order repaired and checked by reading, not compiled, not run on hardware
//
#include <hip/hip_runtime.h>
#include <hip/hip_bf16.h>
#include <math.h>

#define B_    2
#define T_    2048
#define C_    1024
#define NH_   16
#define HS_   64
#define BD_   16
#define DELTA 32
#define MT_   (B_ * T_)          // 4096 rows for big GEMMs
#define ROWS_ (B_ * T_ * NH_)    // 65536 rows for small GEMMs

typedef __attribute__((ext_vector_type(16))) _Float16 v16h;
typedef __attribute__((ext_vector_type(8)))  _Float16 v8h;
typedef __attribute__((ext_vector_type(8)))  float    v8f;
typedef __attribute__((ext_vector_type(2)))  float    v2f;

__device__ __forceinline__ float geluf(float x) {
    return 0.5f * x * (1.0f + erff(x * 0.70710678118654752440f));
}

__device__ __forceinline__ v16h load_afrag(const _Float16* p) {
    // A f16 16x32 fragment: 16 contiguous halves (two adjacent 8-half chunks)
    v8h lo = *(const v8h*)(p);
    v8h hi = *(const v8h*)(p + 8);
    v16h a;
    #pragma unroll
    for (int i = 0; i < 8; ++i) { a[i] = lo[i]; a[i + 8] = hi[i]; }
    return a;
}

// ---------------- elementwise converts ----------------
__global__ void k_cvt(const float* __restrict__ in, _Float16* __restrict__ out, int n) {
    int i = blockIdx.x * blockDim.x + threadIdx.x;
    if (i < n) out[i] = (_Float16)in[i];
}

// in: K x N (f32, row-major) -> out: N x K (f16, row-major)  [weight transpose]
__global__ void k_transpose_cvt(const float* __restrict__ in, _Float16* __restrict__ out,
                                int K, int N) {
    int i = blockIdx.x * blockDim.x + threadIdx.x;
    if (i < K * N) {
        int nn = i / K, kk = i - nn * K;
        out[i] = (_Float16)in[kk * N + nn];
    }
}

// ---------------- big GEMM: C[M,N] = A[M,K] @ B[K,N] + bias ----------------
// A row-major f16, Bt = B transposed (N x K) f16.
// One wave -> one 16x64 strip (4 N-tiles). Explicitly double-buffered:
// loads for K-step i+1 are issued before the 4 WMMAs of K-step i.
__global__ void k_gemm_f16(const _Float16* __restrict__ A, const _Float16* __restrict__ Bt,
                           const float* __restrict__ bias,
                           float* __restrict__ Cf, _Float16* __restrict__ Ch,
                           int M, int N, int K) {
    int wave = (blockIdx.x * blockDim.x + threadIdx.x) >> 5;
    int lane = threadIdx.x & 31;
    int strips_n = N >> 6;                     // 64-wide N strips
    int tm = wave / strips_n;
    int sn = wave - tm * strips_n;
    if (tm >= (M >> 4)) return;

    int r16 = lane & 15;       // A-row / B-col within tile
    int hi  = lane >> 4;       // lane half
    int kgA = hi * 16;         // A f16 16x32: lanes 0-15 own K 0-15; lanes 16-31 own 16-31
    int kgB = hi * 16;         // B f16 32x16: lanes 0-15 own K 0-15; lanes 16-31 own 16-31

    const _Float16* ap  = A  + (size_t)(tm * 16 + r16) * K + kgA;
    const _Float16* br0 = Bt + (size_t)(sn * 64 + 0 * 16 + r16) * K + kgB;
    const _Float16* br1 = Bt + (size_t)(sn * 64 + 1 * 16 + r16) * K + kgB;
    const _Float16* br2 = Bt + (size_t)(sn * 64 + 2 * 16 + r16) * K + kgB;
    const _Float16* br3 = Bt + (size_t)(sn * 64 + 3 * 16 + r16) * K + kgB;

    v8f c0 = {}, c1 = {}, c2 = {}, c3 = {};

    // prologue: fragments for K-step 0
    v16h a  = load_afrag(ap);
    v16h b0 = *(const v16h*)(br0);
    v16h b1 = *(const v16h*)(br1);
    v16h b2 = *(const v16h*)(br2);
    v16h b3 = *(const v16h*)(br3);

    for (int k0 = 32; k0 < K; k0 += 32) {
        // issue next K-step's loads (stay in flight across the WMMAs below)
        v16h an  = load_afrag(ap + k0);
        v16h b0n = *(const v16h*)(br0 + k0);
        v16h b1n = *(const v16h*)(br1 + k0);
        v16h b2n = *(const v16h*)(br2 + k0);
        v16h b3n = *(const v16h*)(br3 + k0);
        __builtin_prefetch(ap + k0 + 128, 0, 1);

        c0 = __builtin_amdgcn_wmma_f32_16x16x32_f16(false, a, false, b0, (short)0, c0,
                                                    false, false);
        c1 = __builtin_amdgcn_wmma_f32_16x16x32_f16(false, a, false, b1, (short)0, c1,
                                                    false, false);
        c2 = __builtin_amdgcn_wmma_f32_16x16x32_f16(false, a, false, b2, (short)0, c2,
                                                    false, false);
        c3 = __builtin_amdgcn_wmma_f32_16x16x32_f16(false, a, false, b3, (short)0, c3,
                                                    false, false);
        a = an; b0 = b0n; b1 = b1n; b2 = b2n; b3 = b3n;
    }
    // epilogue: last K-step
    c0 = __builtin_amdgcn_wmma_f32_16x16x32_f16(false, a, false, b0, (short)0, c0,
                                                false, false);
    c1 = __builtin_amdgcn_wmma_f32_16x16x32_f16(false, a, false, b1, (short)0, c1,
                                                false, false);
    c2 = __builtin_amdgcn_wmma_f32_16x16x32_f16(false, a, false, b2, (short)0, c2,
                                                false, false);
    c3 = __builtin_amdgcn_wmma_f32_16x16x32_f16(false, a, false, b3, (short)0, c3,
                                                false, false);

    #pragma unroll
    for (int j = 0; j < 4; ++j) {
        v8f c = (j == 0) ? c0 : (j == 1) ? c1 : (j == 2) ? c2 : c3;
        int col = sn * 64 + j * 16 + r16;
        float bsv = bias ? bias[col] : 0.0f;
        #pragma unroll
        for (int r = 0; r < 8; ++r) {
            int row = tm * 16 + r + hi * 8;
            float v = c[r] + bsv;
            size_t idx = (size_t)row * N + col;
            if (Ch) Ch[idx] = (_Float16)v;
            else    Cf[idx] = v;
        }
    }
}

// ---------------- small GEMM: Y[M,16] = X[M,16] @ W[16,16], fp32 WMMA ----------------
__global__ void k_small16(const float* __restrict__ X, const float* __restrict__ W,
                          float* __restrict__ Y, int Mtiles) {
    int wave = (blockIdx.x * blockDim.x + threadIdx.x) >> 5;
    int lane = threadIdx.x & 31;
    if (wave >= Mtiles) return;
    int r16 = lane & 15;
    int hi  = lane >> 4;
    int kb  = hi * 2;                   // f32 A 16x4 layout: lanes 0-15 K{0,1}, 16-31 K{2,3}
    const float* xrow = X + (size_t)(wave * 16 + r16) * 16;
    v8f c = {};
    #pragma unroll
    for (int s = 0; s < 4; ++s) {       // K = 16 in four 16x16x4 steps
        v2f a = *(const v2f*)(xrow + s * 4 + kb);
        v2f b;
        b[0] = W[(s * 4 + kb + 0) * 16 + r16];
        b[1] = W[(s * 4 + kb + 1) * 16 + r16];
        c = __builtin_amdgcn_wmma_f32_16x16x4_f32(false, a, false, b, (short)0, c,
                                                  false, false);
    }
    #pragma unroll
    for (int r = 0; r < 8; ++r)
        Y[(size_t)(wave * 16 + r + hi * 8) * 16 + r16] = c[r];
}

// ---------------- sliding-window state: state = mean_valid(Ps[u]) - Ps[t] + b_state ----
__global__ void k_state(const float* __restrict__ Ps, const float* __restrict__ b_state,
                        float* __restrict__ S) {
    int i = blockIdx.x * blockDim.x + threadIdx.x;
    const int total = B_ * T_ * NH_ * BD_;
    if (i >= total) return;
    int k = i & 15;
    int n = (i >> 4) & (NH_ - 1);
    int t = (i >> 8) & (T_ - 1);
    int b = i >> 19;
    int cnt = min(t + 1, DELTA);
    float s = 0.f;
    for (int u = t - cnt + 1; u <= t; ++u)
        s += Ps[(((size_t)b * T_ + u) * NH_ + n) * 16 + k];
    float pst = Ps[(((size_t)b * T_ + t) * NH_ + n) * 16 + k];
    S[i] = s / (float)cnt - pst + b_state[k];
}

// ---------------- U = Pst + Q ; G = R - Pst + b_strain ; E = rel_emb @ w_pos ----------
__global__ void k_prep(const float* __restrict__ Pst, const float* __restrict__ Q,
                       const float* __restrict__ R, const float* __restrict__ b_strain,
                       const float* __restrict__ rel_emb, const float* __restrict__ w_pos,
                       float* __restrict__ U, float* __restrict__ G,
                       float* __restrict__ E) {
    int i = blockIdx.x * blockDim.x + threadIdx.x;
    const int total = B_ * T_ * NH_ * BD_;
    if (i < total) {
        float pst = Pst[i];
        U[i] = pst + Q[i];
        G[i] = R[i] - pst + b_strain[i & 15];
    }
    if (i < DELTA * BD_) {
        int d = i >> 4, k = i & 15;
        float e = 0.f;
        #pragma unroll
        for (int j = 0; j < 16; ++j) e += rel_emb[d * 16 + j] * w_pos[j * 16 + k];
        E[i] = e;
    }
}

// ---------------- windowed attention: one wave per (b,n,t), lane <-> window pos -------
__global__ void k_attn(const float* __restrict__ U, const float* __restrict__ G,
                       const float* __restrict__ E, const float* __restrict__ Pd,
                       const _Float16* __restrict__ valh,
                       const float* __restrict__ w_bond_out,
                       const float* __restrict__ b_bond_out,
                       const float* __restrict__ b_damage,
                       const float* __restrict__ w_damage_out,
                       const float* __restrict__ b_damage_out,
                       _Float16* __restrict__ attn_h) {
    int wave = (blockIdx.x * blockDim.x + threadIdx.x) >> 5;
    int lane = threadIdx.x & 31;
    int t = wave & (T_ - 1);
    int n = (wave >> 11) & (NH_ - 1);
    int b = wave >> 15;
    int d = lane;
    int u = t - DELTA + 1 + d;

    size_t base_t = (((size_t)b * T_ + t) * NH_ + n) * 16;
    float lg;
    if (u >= 0) {
        size_t base_u = (((size_t)b * T_ + u) * NH_ + n) * 16;
        float l  = b_bond_out[0];
        float dm = b_damage_out[0];
        #pragma unroll
        for (int k = 0; k < 16; ++k) {
            float ba = geluf(U[base_u + k] + G[base_t + k] + E[d * 16 + k]);
            l += ba * w_bond_out[k];
            float da = geluf(Pd[base_u + k] - Pd[base_t + k] + b_damage[k]);
            dm += da * w_damage_out[k];
        }
        lg = l - 10.0f / (1.0f + __expf(-dm));   // bond_logit - 10*sigmoid(damage)
    } else {
        lg = -INFINITY;
    }

    // softmax over the 32 window positions (full wave32)
    float m = lg;
    #pragma unroll
    for (int s = 16; s > 0; s >>= 1) m = fmaxf(m, __shfl_xor(m, s, 32));
    float e = (u >= 0) ? __expf(lg - m) : 0.0f;
    float sum = e;
    #pragma unroll
    for (int s = 16; s > 0; s >>= 1) sum += __shfl_xor(sum, s, 32);
    float w = e / sum;

    // out[e] = sum_d w_d * val[u_d, e]; lanes cover the 64 head dims (2 each)
    float a0 = 0.f, a1 = 0.f;
    for (int d2 = 0; d2 < DELTA; ++d2) {
        float wd = __shfl(w, d2, 32);
        int u2 = t - DELTA + 1 + d2;
        if (u2 >= 0) {
            const _Float16* vp = valh + ((size_t)b * T_ + u2) * C_ + n * HS_;
            a0 += wd * (float)vp[lane];
            a1 += wd * (float)vp[lane + 32];
        }
    }
    _Float16* op = attn_h + ((size_t)b * T_ + t) * C_ + n * HS_;
    op[lane]      = (_Float16)a0;
    op[lane + 32] = (_Float16)a1;
}

// ======================================================================================
extern "C" void kernel_launch(void* const* d_in, const int* in_sizes, int n_in,
                              void* d_out, int out_size, void* d_ws, size_t ws_size,
                              hipStream_t stream) {
    (void)in_sizes; (void)n_in; (void)out_size; (void)ws_size;
    const float* x            = (const float*)d_in[0];
    const float* w_disp       = (const float*)d_in[1];
    const float* b_disp       = (const float*)d_in[2];
    const float* w_val        = (const float*)d_in[3];
    const float* b_val        = (const float*)d_in[4];
    const float* rel_emb      = (const float*)d_in[5];
    const float* w_state      = (const float*)d_in[6];
    const float* b_state      = (const float*)d_in[7];
    const float* w_strain     = (const float*)d_in[8];
    const float* b_strain     = (const float*)d_in[9];
    const float* w_state_i    = (const float*)d_in[10];
    const float* w_state_j    = (const float*)d_in[11];
    const float* w_pos        = (const float*)d_in[12];
    const float* w_bond_out   = (const float*)d_in[13];
    const float* b_bond_out   = (const float*)d_in[14];
    const float* w_damage     = (const float*)d_in[15];
    const float* b_damage     = (const float*)d_in[16];
    const float* w_damage_out = (const float*)d_in[17];
    const float* b_damage_out = (const float*)d_in[18];
    const float* w_cproj      = (const float*)d_in[19];
    const float* b_cproj      = (const float*)d_in[20];
    float* out = (float*)d_out;

    char* ws = (char*)d_ws;
    size_t off = 0;
    auto alloc = [&](size_t bytes) -> char* {
        char* p = ws + off;
        off += (bytes + 255) & ~(size_t)255;
        return p;
    };

    _Float16* xh      = (_Float16*)alloc((size_t)MT_ * C_ * 2);     // x in f16
    _Float16* wdispT  = (_Float16*)alloc((size_t)256 * 1024 * 2);   // w_disp^T f16
    _Float16* wvalT   = (_Float16*)alloc((size_t)1024 * 1024 * 2);  // w_val^T f16
    _Float16* wcprojT = (_Float16*)alloc((size_t)1024 * 1024 * 2);  // w_cproj^T f16
    float*    disp    = (float*)   alloc((size_t)MT_ * 256 * 4);    // (b,t,n,k) f32
    _Float16* valh    = (_Float16*)alloc((size_t)MT_ * C_ * 2);     // (b,t,n,e) f16
    float*    Ps      = (float*)   alloc((size_t)ROWS_ * 16 * 4);   // disp@w_state
    float*    Pst     = (float*)   alloc((size_t)ROWS_ * 16 * 4);   // disp@w_strain
    float*    Pd      = (float*)   alloc((size_t)ROWS_ * 16 * 4);   // disp@w_damage
    float*    state   = (float*)   alloc((size_t)ROWS_ * 16 * 4);
    float*    Q       = (float*)   alloc((size_t)ROWS_ * 16 * 4);   // state@w_state_j
    float*    R       = (float*)   alloc((size_t)ROWS_ * 16 * 4);   // state@w_state_i
    float*    Ubuf    = (float*)   alloc((size_t)ROWS_ * 16 * 4);
    float*    Gbuf    = (float*)   alloc((size_t)ROWS_ * 16 * 4);
    float*    Ebuf    = (float*)   alloc((size_t)DELTA * 16 * 4);
    _Float16* attn_h  = (_Float16*)alloc((size_t)MT_ * C_ * 2);

    const int THR = 256;

    // 1) converts / weight transposes
    k_cvt<<<(MT_ * C_) / THR, THR, 0, stream>>>(x, xh, MT_ * C_);
    k_transpose_cvt<<<(256 * 1024) / THR, THR, 0, stream>>>(w_disp, wdispT, 1024, 256);
    k_transpose_cvt<<<(1024 * 1024) / THR, THR, 0, stream>>>(w_val, wvalT, 1024, 1024);
    k_transpose_cvt<<<(1024 * 1024) / THR, THR, 0, stream>>>(w_cproj, wcprojT, 1024, 1024);

    // 2) big projections (f16 WMMA, f32 accumulate), 16x64 strip per wave
    k_gemm_f16<<<(MT_ / 16) * (256 / 64) / 8, THR, 0, stream>>>(
        xh, wdispT, b_disp, disp, (_Float16*)nullptr, MT_, 256, C_);
    k_gemm_f16<<<(MT_ / 16) * (1024 / 64) / 8, THR, 0, stream>>>(
        xh, wvalT, b_val, (float*)nullptr, valh, MT_, 1024, C_);

    // 3) flat 16-wide projections of disp (fp32 WMMA 16x16x4)
    const int MT16 = ROWS_ / 16;                 // 4096 tiles
    k_small16<<<MT16 / 8, THR, 0, stream>>>(disp, w_state, Ps, MT16);
    k_small16<<<MT16 / 8, THR, 0, stream>>>(disp, w_strain, Pst, MT16);
    k_small16<<<MT16 / 8, THR, 0, stream>>>(disp, w_damage, Pd, MT16);

    // 4) sliding-window state, then its projections
    k_state<<<(ROWS_ * 16) / THR, THR, 0, stream>>>(Ps, b_state, state);
    k_small16<<<MT16 / 8, THR, 0, stream>>>(state, w_state_j, Q, MT16);
    k_small16<<<MT16 / 8, THR, 0, stream>>>(state, w_state_i, R, MT16);

    // 5) fuse additive terms: U = Pst+Q, G = R-Pst+b_strain, E = rel_emb@w_pos
    k_prep<<<(ROWS_ * 16) / THR, THR, 0, stream>>>(Pst, Q, R, b_strain, rel_emb, w_pos,
                                                   Ubuf, Gbuf, Ebuf);

    // 6) windowed attention (one wave per (b,n,t))
    k_attn<<<(B_ * NH_ * T_) / 8, THR, 0, stream>>>(Ubuf, Gbuf, Ebuf, Pd, valh,
                                                    w_bond_out, b_bond_out, b_damage,
                                                    w_damage_out, b_damage_out, attn_h);

    // 7) output projection (f16 WMMA), bias, f32 out
    k_gemm_f16<<<(MT_ / 16) * (1024 / 64) / 8, THR, 0, stream>>>(
        attn_h, wcprojT, b_cproj, out, (_Float16*)nullptr, MT_, 1024, C_);
}
